// GNBlock_5076651344276
// MI455X (gfx1250) — matrix operand order repaired and checked by reading, not compiled, not run on hardware
//
#include <hip/hip_runtime.h>
#include <hip/hip_bf16.h>
#include <cmath>

// ---------------------------------------------------------------------------
// GN block for MI455X (gfx1250): bf16 WMMA edge/attention MLPs with the
// broadcast parts of the K-reduction hoisted into per-node precomputes.
// TDM (tensor_load_to_lds) double-buffered on TENSORcnt for the edge stream.
//   B=8, N=256, Dn=De=32, H=64 (per net), fused hidden HH=128, ctxt=48.
// ---------------------------------------------------------------------------

#define GN_B   8
#define GN_N   256
#define GN_D   32
#define GN_H   64
#define GN_HH  128

typedef __attribute__((ext_vector_type(16))) __bf16 v16bf;
typedef __attribute__((ext_vector_type(8)))  float  v8f;
typedef __attribute__((ext_vector_type(4)))  unsigned int u32x4;
typedef __attribute__((ext_vector_type(8)))  int i32x8;
typedef __attribute__((ext_vector_type(4)))  int i32x4;

struct F8  { float f[8]; };
struct S16 { unsigned short s[16]; };

#if defined(__has_builtin)
#  if __has_builtin(__builtin_amdgcn_tensor_load_to_lds)
#    define GN_HAVE_TDM 1
#  endif
#  if __has_builtin(__builtin_amdgcn_s_wait_tensorcnt)
#    define GN_HAVE_TWAIT 1
#  endif
#endif
#ifndef GN_HAVE_TDM
#  define GN_HAVE_TDM 0
#endif
#ifndef GN_HAVE_TWAIT
#  define GN_HAVE_TWAIT 0
#endif

#if defined(__HIP_DEVICE_COMPILE__)
__device__ __forceinline__ float fast_rcp(float x) { return __builtin_amdgcn_rcpf(x); }
#else
__device__ __forceinline__ float fast_rcp(float x) { return 1.0f / x; }
#endif

// SiLU with v_rcp_f32 instead of the IEEE divide expansion.
__device__ __forceinline__ float silu_f(float x) {
  return x * fast_rcp(1.0f + __expf(-x));
}
// softplus with v_log_f32 instead of log1pf's libm expansion.
__device__ __forceinline__ float softplus_f(float x) {
  return (x > 20.0f) ? x : __logf(1.0f + __expf(x));
}
__device__ __forceinline__ unsigned short f2bf(float x) {
  __bf16 h = (__bf16)x;
  return __builtin_bit_cast(unsigned short, h);
}

#if defined(__HIP_DEVICE_COMPILE__)
__device__ __forceinline__ v8f wmma_bf16(v16bf a, v16bf b, v8f c) {
  // D = A(16x32 bf16) x B(32x16 bf16) + C(16x16 f32)
  return __builtin_amdgcn_wmma_f32_16x16x32_bf16(false, a, false, b, (short)0, c,
                                                 false, false);
}
#else
__device__ __forceinline__ v8f wmma_bf16(v16bf, v16bf, v8f c) { return c; }
#endif

// Load one 32x16 bf16 B-operand laid out per-lane (32 contiguous bytes/lane).
__device__ __forceinline__ v16bf load_bop(const unsigned short* __restrict__ p,
                                          int op, int lane) {
  const uint4* q = reinterpret_cast<const uint4*>(p) + (size_t)(op * 32 + lane) * 2;
  S16 u;
  *reinterpret_cast<uint4*>(&u.s[0]) = q[0];
  *reinterpret_cast<uint4*>(&u.s[8]) = q[1];
  return __builtin_bit_cast(v16bf, u);
}

// ---------------------------------------------------------------------------
// Prep kernels
// ---------------------------------------------------------------------------

__global__ void gn_zero_pooled(float* pooled) {
  pooled[blockIdx.x * 256 + threadIdx.x] = 0.0f;
}

// Pack W1e (edges slice of fused layer-1 weight, rows 64..95 of 144) into the
// per-lane bf16 B-operand layout: lane l holds column (l&15), K = 16*(l>>4)+q.
__global__ void gn_pack_w1(const float* __restrict__ We1,
                           const float* __restrict__ Wa1,
                           unsigned short* __restrict__ w1e) {
  int idx = blockIdx.x * 256 + threadIdx.x;
  if (idx >= 8 * 32 * 16) return;
  int q = idx & 15, l = (idx >> 4) & 31, t = idx >> 9;   // t: hidden N-tile 0..7
  int K   = 16 * (l >> 4) + q;                           // 0..31 (edge feature)
  int col = t * 16 + (l & 15);                           // fused hidden 0..127
  int r   = 64 + K;                                      // row in 144-wide input
  float v = (col < 64) ? We1[r * 64 + col] : Wa1[r * 64 + (col - 64)];
  w1e[idx] = f2bf(v);
}

// Pack layer-2 weights. Operand op = n*2 + c:  n in {0,1}: We2 cols 16n..16n+15,
// n==2: Wa2 in column 0; c = K-chunk (K = 32c .. 32c+31).
__global__ void gn_pack_w2(const float* __restrict__ We2,
                           const float* __restrict__ Wa2,
                           unsigned short* __restrict__ w2p) {
  int idx = blockIdx.x * 256 + threadIdx.x;
  if (idx >= 6 * 32 * 16) return;
  int q = idx & 15, l = (idx >> 4) & 31, op = idx >> 9;
  int n = op >> 1, c = op & 1;
  int K = 32 * c + 16 * (l >> 4) + q;                    // 0..63 (hidden)
  float v;
  if (n < 2) v = We2[K * 32 + n * 16 + (l & 15)];
  else       v = ((l & 15) == 0) ? Wa2[K] : 0.0f;
  w2p[idx] = f2bf(v);
}

// Per-node precomputes over the fused 128-wide hidden:
//   T(b,i,h) = sum_k nodes*W1[send] + sum_k ctxt*W1[ctxt] + bias1
//   R(b,j,h) = sum_k nodes*W1[recv]
__global__ void gn_prep_tr(const float* __restrict__ nodes,
                           const float* __restrict__ globs,
                           const float* __restrict__ cndts,
                           const float* __restrict__ We1, const float* __restrict__ be1,
                           const float* __restrict__ Wa1, const float* __restrict__ ba1,
                           float* __restrict__ Tb, float* __restrict__ Rb) {
  int idx = blockIdx.x * 256 + threadIdx.x;              // 2 * 2048 * 128
  int sel = idx >> 18;
  int r = idx & 262143;
  int h = r & 127;
  int n = r >> 7;                                        // flat node 0..2047
  int b = n >> 8;
  int col = h & 63;
  const float* W  = (h < 64) ? We1 : Wa1;
  const float* nr = nodes + (size_t)n * 32;
  if (sel == 0) {
    float acc = (h < 64) ? be1[col] : ba1[col];
    for (int k = 0; k < 32; ++k) acc += nr[k] * W[k * 64 + col];
    for (int k = 0; k < 32; ++k) acc += globs[b * 32 + k] * W[(96 + k) * 64 + col];
    for (int k = 0; k < 16; ++k) acc += cndts[b * 16 + k] * W[(128 + k) * 64 + col];
    Tb[r] = acc;
  } else {
    float acc = 0.0f;
    for (int k = 0; k < 32; ++k) acc += nr[k] * W[(32 + k) * 64 + col];
    Rb[r] = acc;
  }
}

// ---------------------------------------------------------------------------
// Main fused edge + attention + pooled reduction kernel.
// grid = B * (N/16) receiver tiles, 256 threads = 8 waves, wave w owns
// senders i = w, w+8, ... (32 iterations), TDM double-buffered.
// ---------------------------------------------------------------------------

__global__ __launch_bounds__(256, 1)
void gn_edge_kernel(const float* __restrict__ edges, const int* __restrict__ adj,
                    const float* __restrict__ Tb, const float* __restrict__ Rb,
                    const unsigned short* __restrict__ w1e,
                    const unsigned short* __restrict__ w2p,
                    const float* __restrict__ be2, const float* __restrict__ ba2,
                    float* __restrict__ pooled) {
  __shared__ float Rt[16 * GN_HH];                                       // 8 KB
  __shared__ __attribute__((aligned(16))) float stagef[8][2][16 * GN_D]; // 32 KB (TDM dst)
  __shared__ __attribute__((aligned(16))) unsigned short h2s[8][16 * GN_H]; // 16 KB
  __shared__ float attnst[8][16];

  const int tid  = threadIdx.x;
  const int w    = tid >> 5;
  const int lane = tid & 31;
  const int lo   = lane & 15;
  const int hi   = lane >> 4;
  const int b    = blockIdx.x >> 4;
  const int j0   = (blockIdx.x & 15) << 4;

  // R tile for the 16 receivers of this block.
  for (int q = tid; q < 16 * GN_HH; q += 256) {
    int r = q >> 7, h = q & 127;
    Rt[q] = Rb[(size_t)(b * GN_N + j0 + r) * GN_HH + h];
  }
  __syncthreads();

  // Layer-1 B operands stay resident in VGPRs for the whole i-loop.
  v16bf b1[8];
#pragma unroll
  for (int t = 0; t < 8; ++t) b1[t] = load_bop(w1e, t, lane);

  // Receiver bias tiles (R) resident in registers for the whole i-loop.
  float rbasef[8][8];
#pragma unroll
  for (int t = 0; t < 8; ++t)
#pragma unroll
    for (int v = 0; v < 8; ++v)
      rbasef[t][v] = Rt[(v + 8 * hi) * GN_HH + t * 16 + lo];

  float be2v[2];
  be2v[0] = be2[lo];
  be2v[1] = be2[16 + lo];
  const float ba2v = (lo == 0) ? ba2[0] : 0.0f;

  float accf[2][8];
#pragma unroll
  for (int n = 0; n < 2; ++n)
#pragma unroll
    for (int v = 0; v < 8; ++v) accf[n][v] = 0.0f;

#if GN_HAVE_TDM
  // Issue a TDM load of edges[b, i, j0:j0+16, 0:32] (f32, 2 KB) into buf.
  auto issue_tdm = [&](int i, int buf) {
    const float* ebase = edges + ((size_t)(b * GN_N + i) * GN_N + j0) * GN_D;
    unsigned long long gaddr = (unsigned long long)(const void*)ebase;
    unsigned int ldsoff = (unsigned int)(unsigned long long)(const void*)&stagef[w][buf][0];
    unsigned g0w0 = 1u;  // count=1, gather off
    unsigned g0w1 = (unsigned)__builtin_amdgcn_readfirstlane((int)ldsoff);
    unsigned g0w2 = (unsigned)__builtin_amdgcn_readfirstlane((int)(unsigned)gaddr);
    unsigned g0w3 =
        ((unsigned)__builtin_amdgcn_readfirstlane((int)((unsigned)(gaddr >> 32) & 0x1FFFFFFu))) |
        (2u << 30);  // type = "image"
    u32x4 g0 = {g0w0, g0w1, g0w2, g0w3};
    // group1: data_size=4B, tensor_dim0=32, tensor_dim1=16, tile 32x16, stride0=32
    i32x8 g1 = {(int)(2u << 16), (int)(32u << 16), (int)(16u << 16), (int)(32u << 16),
                16, 32, 0, 0};
    i32x4 gz = {0, 0, 0, 0};
#if __clang_major__ >= 23
    i32x8 gz8 = {0, 0, 0, 0, 0, 0, 0, 0};
    __builtin_amdgcn_tensor_load_to_lds(g0, g1, gz, gz, gz8, 0);
#else
    __builtin_amdgcn_tensor_load_to_lds(g0, g1, gz, gz, 0);
#endif
  };
  issue_tdm(w, 0);  // prologue: first tile in flight
#endif

  int p = 0;
  for (int i = w; i < GN_N; i += 8) {
    const bool hasnext = (i + 8 < GN_N);
    if (hasnext)
      __builtin_prefetch(edges + ((size_t)(b * GN_N + i + 8) * GN_N + j0) * GN_D, 0, 0);

#if GN_HAVE_TDM
    // Double buffer: issue next tile, then wait until only it is outstanding.
    if (hasnext) {
      issue_tdm(i + 8, p ^ 1);
#if GN_HAVE_TWAIT
      __builtin_amdgcn_s_wait_tensorcnt(1);
#else
      asm volatile("s_wait_tensorcnt 0x1" ::: "memory");
#endif
    } else {
#if GN_HAVE_TWAIT
      __builtin_amdgcn_s_wait_tensorcnt(0);
#else
      asm volatile("s_wait_tensorcnt 0x0" ::: "memory");
#endif
    }
    asm volatile("" ::: "memory");
#else
    {
      const float* ebase = edges + ((size_t)(b * GN_N + i) * GN_N + j0) * GN_D;
      for (int q = 0; q < 16; ++q) stagef[w][p][q * 32 + lane] = ebase[q * 32 + lane];
      asm volatile("" ::: "memory");
    }
#endif

    // Sender+ctxt+bias tile values (depend on column only).
    float tv[8];
    const float* Trow = Tb + (size_t)(b * GN_N + i) * GN_HH;
#pragma unroll
    for (int t = 0; t < 8; ++t) tv[t] = Trow[t * 16 + lo];

    // ---- build A (16 j-rows x 32 edge feats, bf16 A-layout) ----
    const float* st = &stagef[w][p][0];
    const int k0 = hi ? 16 : 0;
    const float4* ap = reinterpret_cast<const float4*>(st + lo * GN_D + k0);
    float4 f0 = ap[0], f1 = ap[1], f2 = ap[2], f3 = ap[3];
    S16 a1s;
    a1s.s[0] = f2bf(f0.x);  a1s.s[1] = f2bf(f0.y);  a1s.s[2]  = f2bf(f0.z);  a1s.s[3]  = f2bf(f0.w);
    a1s.s[4] = f2bf(f1.x);  a1s.s[5] = f2bf(f1.y);  a1s.s[6]  = f2bf(f1.z);  a1s.s[7]  = f2bf(f1.w);
    a1s.s[8] = f2bf(f2.x);  a1s.s[9] = f2bf(f2.y);  a1s.s[10] = f2bf(f2.z);  a1s.s[11] = f2bf(f2.w);
    a1s.s[12] = f2bf(f3.x); a1s.s[13] = f2bf(f3.y); a1s.s[14] = f2bf(f3.z);  a1s.s[15] = f2bf(f3.w);
    v16bf A1 = __builtin_bit_cast(v16bf, a1s);

    F8 oe[2];
    F8 oa;

#pragma unroll
    for (int halfsel = 0; halfsel < 2; ++halfsel) {
      // Layer 1: 4 hidden tiles (edge net or attn net), C = T + R bias.
#pragma unroll
      for (int tt = 0; tt < 4; ++tt) {
        int t = halfsel * 4 + tt;
        F8 c;
#pragma unroll
        for (int v = 0; v < 8; ++v) c.f[v] = rbasef[t][v] + tv[t];
        v8f h = wmma_bf16(A1, b1[t], __builtin_bit_cast(v8f, c));
        F8 hf = __builtin_bit_cast(F8, h);
#pragma unroll
        for (int v = 0; v < 8; ++v) {
          float x = silu_f(hf.f[v]);
          h2s[w][(v + 8 * hi) * GN_H + tt * 16 + lo] = f2bf(x);
        }
      }
      asm volatile("" ::: "memory");  // order LDS store->load (wave-private, HW in-order)

      // Re-layout hidden C->A via LDS (two K=32 chunks).
      v16bf A2[2];
#pragma unroll
      for (int c = 0; c < 2; ++c) {
        const unsigned short* hp = &h2s[w][0] + lo * GN_H + c * 32 + hi * 16;
        const uint4* q = reinterpret_cast<const uint4*>(hp);
        S16 u;
        *reinterpret_cast<uint4*>(&u.s[0]) = q[0];  // K = c*32 + hi*16 + 0..7
        *reinterpret_cast<uint4*>(&u.s[8]) = q[1];  // K = c*32 + hi*16 + 8..15
        A2[c] = __builtin_bit_cast(v16bf, u);
      }

      if (halfsel == 0) {
        // Layer 2 edge net: delta_e = hidden_e[16x64] x We2[64x32] + be2
#pragma unroll
        for (int n = 0; n < 2; ++n) {
          F8 c;
#pragma unroll
          for (int v = 0; v < 8; ++v) c.f[v] = be2v[n];
          v8f o = wmma_bf16(A2[0], load_bop(w2p, n * 2 + 0, lane),
                            __builtin_bit_cast(v8f, c));
          o = wmma_bf16(A2[1], load_bop(w2p, n * 2 + 1, lane), o);
          oe[n] = __builtin_bit_cast(F8, o);
        }
      } else {
        // Layer 2 attn net: a = hidden_a[16x64] x Wa2[64x1] + ba2 (column 0)
        F8 c;
#pragma unroll
        for (int v = 0; v < 8; ++v) c.f[v] = ba2v;
        v8f o = wmma_bf16(A2[0], load_bop(w2p, 4, lane), __builtin_bit_cast(v8f, c));
        o = wmma_bf16(A2[1], load_bop(w2p, 5, lane), o);
        oa = __builtin_bit_cast(F8, o);
      }
    }

    // ---- per-row softplus(attn)*mask, broadcast via LDS ----
    asm volatile("" ::: "memory");
    if (lo == 0) {  // lanes 0 (rows 0..7) and 16 (rows 8..15) hold column 0
      const int* arow = adj + (size_t)(b * GN_N + i) * GN_N + j0;
#pragma unroll
      for (int v = 0; v < 8; ++v) {
        int r = v + 8 * hi;
        attnst[w][r] = arow[r] ? softplus_f(oa.f[v]) : 0.0f;
      }
    }
    asm volatile("" ::: "memory");
    float wv[8];
#pragma unroll
    for (int v = 0; v < 8; ++v) wv[v] = attnst[w][v + 8 * hi];

    // pooled partial: (edges + delta) * weight   (mask folded into weight)
#pragma unroll
    for (int n = 0; n < 2; ++n)
#pragma unroll
      for (int v = 0; v < 8; ++v) {
        float ev = st[(v + 8 * hi) * GN_D + n * 16 + lo];
        accf[n][v] += (ev + oe[n].f[v]) * wv[v];
      }

    p ^= 1;
  }

  // Flush partials: hardware f32 atomics into pooled[b, j0+row, col].
#pragma unroll
  for (int n = 0; n < 2; ++n)
#pragma unroll
    for (int v = 0; v < 8; ++v) {
      int r = v + 8 * hi;
      unsafeAtomicAdd(&pooled[((size_t)(b * GN_N) + j0 + r) * GN_D + n * 16 + lo],
                      accf[n][v]);
    }
}

// ---------------------------------------------------------------------------
// Node block (tiny: ~38 MF) — plain VALU.
// ---------------------------------------------------------------------------

__global__ void gn_node1(const float* __restrict__ pooled, const float* __restrict__ nodes,
                         const float* __restrict__ globs, const float* __restrict__ cndts,
                         const float* __restrict__ Wn1, const float* __restrict__ bn1,
                         float* __restrict__ hidn) {
  int idx = blockIdx.x * 256 + threadIdx.x;  // 2048*64
  int h = idx & 63;
  int n = idx >> 6;
  int b = n >> 8;
  const float* pr = pooled + (size_t)n * 32;
  const float* nr = nodes + (size_t)n * 32;
  float acc = bn1[h];
  for (int k = 0; k < 32; ++k) acc += pr[k] * Wn1[k * 64 + h];
  for (int k = 0; k < 32; ++k) acc += nr[k] * Wn1[(32 + k) * 64 + h];
  for (int k = 0; k < 32; ++k) acc += globs[b * 32 + k] * Wn1[(64 + k) * 64 + h];
  for (int k = 0; k < 16; ++k) acc += cndts[b * 16 + k] * Wn1[(96 + k) * 64 + h];
  hidn[idx] = silu_f(acc);
}

__global__ void gn_node2(const float* __restrict__ hidn, const float* __restrict__ nodes,
                         const float* __restrict__ Wn2, const float* __restrict__ bn2,
                         float* __restrict__ out) {
  int idx = blockIdx.x * 256 + threadIdx.x;  // 2048*32
  int d = idx & 31;
  int n = idx >> 5;
  float acc = bn2[d];
  const float* hr = hidn + (size_t)n * 64;
  for (int h = 0; h < 64; ++h) acc += hr[h] * Wn2[h * 32 + d];
  out[idx] = nodes[idx] + acc;
}

// ---------------------------------------------------------------------------

extern "C" void kernel_launch(void* const* d_in, const int* in_sizes, int n_in,
                              void* d_out, int out_size, void* d_ws, size_t ws_size,
                              hipStream_t stream) {
  const float* nodes = (const float*)d_in[0];
  const float* edges = (const float*)d_in[1];
  const float* globs = (const float*)d_in[2];
  const float* cndts = (const float*)d_in[3];
  const int*   adj   = (const int*)d_in[4];
  const float* We1 = (const float*)d_in[5];
  const float* be1 = (const float*)d_in[6];
  const float* We2 = (const float*)d_in[7];
  const float* be2 = (const float*)d_in[8];
  const float* Wa1 = (const float*)d_in[9];
  const float* ba1 = (const float*)d_in[10];
  const float* Wa2 = (const float*)d_in[11];
  const float* ba2 = (const float*)d_in[12];
  const float* Wn1 = (const float*)d_in[13];
  const float* bn1 = (const float*)d_in[14];
  const float* Wn2 = (const float*)d_in[15];
  const float* bn2 = (const float*)d_in[16];

  char* ws = (char*)d_ws;
  float* pooled         = (float*)(ws + 0);              // 262144 B
  float* Tb             = (float*)(ws + 262144);         // 1 MiB
  float* Rb             = (float*)(ws + 1310720);        // 1 MiB
  float* hidn           = (float*)(ws + 2359296);        // 512 KiB
  unsigned short* w1e   = (unsigned short*)(ws + 2883584);  // 8 KiB
  unsigned short* w2p   = (unsigned short*)(ws + 2891776);  // 6 KiB

  gn_zero_pooled<<<256, 256, 0, stream>>>(pooled);
  gn_pack_w1<<<16, 256, 0, stream>>>(We1, Wa1, w1e);
  gn_pack_w2<<<12, 256, 0, stream>>>(We2, Wa2, w2p);
  gn_prep_tr<<<2048, 256, 0, stream>>>(nodes, globs, cndts, We1, be1, Wa1, ba1, Tb, Rb);
  gn_edge_kernel<<<GN_B * (GN_N / 16), 256, 0, stream>>>(edges, adj, Tb, Rb, w1e, w2p,
                                                         be2, ba2, pooled);
  gn_node1<<<512, 256, 0, stream>>>(pooled, nodes, globs, cndts, Wn1, bn1, hidn);
  gn_node2<<<256, 256, 0, stream>>>(hidn, nodes, Wn2, bn2, (float*)d_out);
}
